// Attention_7602092114523
// MI455X (gfx1250) — matrix operand order repaired and checked by reading, not compiled, not run on hardware
//
#include <hip/hip_runtime.h>
#include <hip/hip_bf16.h>

#define B_ 2
#define N_ 2048
#define C_ 1024
#define H_ 16
#define D_ 64
#define SCALE_ 0.125f   // 64^-0.5

typedef __attribute__((ext_vector_type(16))) __bf16 v16bf;
typedef __attribute__((ext_vector_type(8)))  __bf16 v8bf;
typedef __attribute__((ext_vector_type(8)))  float  v8f;

#if defined(__has_builtin)
#  if __has_builtin(__builtin_amdgcn_tensor_load_to_lds)
#    define HAVE_TDM 1
#  endif
#endif
#ifndef HAVE_TDM
#  define HAVE_TDM 0
#endif

static __device__ __forceinline__ v8f wmma_bf16(v16bf a, v16bf b, v8f c) {
  return __builtin_amdgcn_wmma_f32_16x16x32_bf16(false, a, false, b, (short)0, c,
                                                 false, false);
}

static __device__ __forceinline__ v16bf make_a(v8bf lo, v8bf hi) {
  v16bf a;
#pragma unroll
  for (int i = 0; i < 8; ++i) { a[i] = lo[i]; a[i + 8] = hi[i]; }
  return a;
}

#if HAVE_TDM
typedef __attribute__((ext_vector_type(4))) unsigned int u32x4_t;
typedef __attribute__((ext_vector_type(8))) int i32x8_t;
typedef __attribute__((ext_vector_type(4))) int i32x4_t;

// Generic LDS pointer: low 32 bits of the flat address are the LDS byte addr.
static __device__ __forceinline__ unsigned lds_addr_of(const void* p) {
  return (unsigned)(unsigned long long)p;
}

// 2D TDM tile load: tile_d0 contiguous elements per row, tile_d1 rows,
// rows separated by row_stride elements; bf16 (2-byte) elements.
// D# layout per CDNA5 ISA 8.3/8.4 (group0: count/lds/global/type,
// group1: data_size + dims + strides).  Groups 2/3 (and the trailing group)
// zero for <=2D tensors.  This toolchain's builtin takes 6 args (clang-23
// form: g0 u32x4, g1 i32x8, g2 i32x4, g3 i32x4, extra i32x8, cpol).
static __device__ __forceinline__ void tdm_load_tile_2d(
    unsigned lds_off, const void* gptr, unsigned tile_d0, unsigned tile_d1,
    unsigned row_stride) {
  unsigned long long ga = (unsigned long long)gptr;
  u32x4_t g0;
  g0[0] = 1u;                                       // count=1, user, no gather
  g0[1] = lds_off;                                  // lds_addr (bytes)
  g0[2] = (unsigned)ga;                             // global_addr[31:0]
  g0[3] = (unsigned)((ga >> 32) & 0x01FFFFFFu) | (2u << 30);  // [56:32]|type=2
  i32x8_t g1;
  g1[0] = (int)(1u << 16);                          // wg_mask=0, data_size=2B
  g1[1] = (int)((tile_d0 & 0xFFFFu) << 16);         // tensor_dim0 lo16
  g1[2] = (int)((tile_d0 >> 16) | ((tile_d1 & 0xFFFFu) << 16)); // d0 hi|d1 lo
  g1[3] = (int)((tile_d1 >> 16) | ((tile_d0 & 0xFFFFu) << 16)); // d1 hi|tile0
  g1[4] = (int)(tile_d1 & 0xFFFFu);                 // tile_dim1, tile_dim2=0
  g1[5] = (int)row_stride;                          // tensor_dim0_stride lo32
  g1[6] = 0;
  g1[7] = 0;
  i32x4_t z4 = {0, 0, 0, 0};
  i32x8_t z8 = {0, 0, 0, 0, 0, 0, 0, 0};
  __builtin_amdgcn_tensor_load_to_lds(g0, g1, z4, z4, z8, 0);
}
#endif

// ---------------------------------------------------------------- pre-passes
__global__ void k_convert(const float* __restrict__ in, __bf16* __restrict__ out,
                          int n) {
  int i = blockIdx.x * blockDim.x + threadIdx.x;
  if (i < n) out[i] = (__bf16)in[i];
}

__global__ void k_transpose_cvt(const float* __restrict__ in,
                                __bf16* __restrict__ out, int rows, int cols) {
  int i = blockIdx.x * blockDim.x + threadIdx.x;
  if (i < rows * cols) {
    int r = i / cols, c = i % cols;
    out[(size_t)c * rows + r] = (__bf16)in[i];
  }
}

// ---------------------------------------------------------------- QKV GEMM
// 64x64 tile per wave (4x4 grid of 16x16 WMMA tiles): each A/B fragment is
// reused 4x -> ~32 flops/byte of L2 traffic.
__global__ void __launch_bounds__(128)
k_qkv_gemm(const __bf16* __restrict__ xb, const __bf16* __restrict__ wT,
           __bf16* __restrict__ Qb, __bf16* __restrict__ Kb,
           __bf16* __restrict__ Vt) {
  const int lane = threadIdx.x & 31;
  const int wave = threadIdx.x >> 5;
  const int ln = lane & 15, hf = lane >> 4;
  const int NTB = (3 * C_) / 64;               // 48 column blocks
  int wtile = blockIdx.x * 4 + wave;           // 0 .. 64*48-1
  int ntb = wtile % NTB, mtb = wtile / NTB;

  const __bf16* ap0 = xb + (size_t)(mtb * 64 + ln) * C_ + hf * 16;
  const __bf16* bp0 = wT + (size_t)(ntb * 64 + ln) * C_ + hf * 16;

  v8f acc[4][4];
#pragma unroll
  for (int i = 0; i < 4; ++i)
#pragma unroll
    for (int j = 0; j < 4; ++j) acc[i][j] = v8f{};

  for (int k0 = 0; k0 < C_; k0 += 32) {
    v16bf a[4], b[4];
#pragma unroll
    for (int i = 0; i < 4; ++i)
      a[i] = make_a(*(const v8bf*)(ap0 + (size_t)i * 16 * C_ + k0),
                    *(const v8bf*)(ap0 + (size_t)i * 16 * C_ + k0 + 8));
#pragma unroll
    for (int j = 0; j < 4; ++j)
      b[j] = *(const v16bf*)(bp0 + (size_t)j * 16 * C_ + k0);
#pragma unroll
    for (int i = 0; i < 4; ++i)
#pragma unroll
      for (int j = 0; j < 4; ++j) acc[i][j] = wmma_bf16(a[i], b[j], acc[i][j]);
  }

  // scatter into Q*scale / K / V^T.  64-col block never crosses an s or h
  // boundary, so s/hh are wave-uniform; d = j*16 + ln.
  int jb = ntb * 64;
  int s = jb >> 10;
  int hh = (jb >> 6) & (H_ - 1);
#pragma unroll
  for (int j = 0; j < 4; ++j) {
    int d = j * 16 + ln;
#pragma unroll
    for (int i = 0; i < 4; ++i)
#pragma unroll
      for (int r = 0; r < 8; ++r) {
        int t = mtb * 64 + i * 16 + r + hf * 8;
        int bb = t >> 11, n = t & (N_ - 1);
        float v = acc[i][j][r];
        if (s == 0)
          Qb[((size_t)(bb * H_ + hh) * N_ + n) * D_ + d] = (__bf16)(v * SCALE_);
        else if (s == 1)
          Kb[((size_t)(bb * H_ + hh) * N_ + n) * D_ + d] = (__bf16)v;
        else
          Vt[((size_t)(bb * H_ + hh) * D_ + d) * N_ + n] = (__bf16)v;
      }
  }
}

// ---------------------------------------------------------------- attention
// Block = 4 waves on the same (b,h), 64 consecutive queries (16 per wave).
// K/V 32-key tiles are staged once per block into double-buffered LDS via the
// Tensor Data Mover (fallback: cooperative sync copy), overlapped with WMMA.
__global__ void __launch_bounds__(128)
k_attention(const __bf16* __restrict__ Qb, const __bf16* __restrict__ Kb,
            const __bf16* __restrict__ Vt, __bf16* __restrict__ attn_out) {
  __shared__ __align__(32) __bf16 KbufS[2][32][64];   // [key][d]
  __shared__ __align__(32) __bf16 VbufS[2][64][32];   // [d][key]
  __shared__ __align__(32) __bf16 Pbuf[4][16][32];    // per-wave P tile

  const int tid = threadIdx.x;
  const int lane = tid & 31;
  const int wave = tid >> 5;
  const int ln = lane & 15, hf = lane >> 4;

  int bh = blockIdx.x >> 5;                   // 32 blocks per (b,h)
  int h = bh & (H_ - 1), b = bh >> 4;
  int nq0 = (blockIdx.x & 31) * 64 + wave * 16;

  const __bf16* Q = Qb + (size_t)(b * H_ + h) * N_ * D_;
  const __bf16* K = Kb + (size_t)(b * H_ + h) * N_ * D_;
  const __bf16* V = Vt + (size_t)(b * H_ + h) * D_ * N_;

  const __bf16* qrow = Q + (size_t)(nq0 + ln) * D_;
  v16bf aQ0 = make_a(*(const v8bf*)(qrow + hf * 16),
                     *(const v8bf*)(qrow + 8 + hf * 16));
  v16bf aQ1 = make_a(*(const v8bf*)(qrow + 32 + hf * 16),
                     *(const v8bf*)(qrow + 40 + hf * 16));

  float mrow[8], lrow[8];
  v8f o0 = {}, o1 = {}, o2 = {}, o3 = {};
#pragma unroll
  for (int r = 0; r < 8; ++r) { mrow[r] = -3.0e38f; lrow[r] = 0.0f; }

  __bf16* pp = &Pbuf[wave][0][0];

  // ---- tile staging (TDM or cooperative fallback)
  auto stage = [&](int bufi, int kt) {
#if HAVE_TDM
    if (wave == 0)
      tdm_load_tile_2d(lds_addr_of(&KbufS[bufi][0][0]),
                       K + (size_t)kt * D_, D_, 32, D_);
    else if (wave == 1)
      tdm_load_tile_2d(lds_addr_of(&VbufS[bufi][0][0]),
                       V + kt, 32, D_, N_);
#else
    {
      int row = tid >> 2, seg = tid & 3;      // 32 rows x 4 segs of 16 bf16
      *(v16bf*)(&KbufS[bufi][row][seg * 16]) =
          *(const v16bf*)(K + (size_t)(kt + row) * D_ + seg * 16);
    }
    {
      int row = tid >> 1, seg = tid & 1;      // 64 rows x 2 segs of 16 bf16
      *(v16bf*)(&VbufS[bufi][row][seg * 16]) =
          *(const v16bf*)(V + (size_t)row * N_ + kt + seg * 16);
    }
#endif
  };

  stage(0, 0);
  int p = 0;
  for (int kt = 0; kt < N_; kt += 32) {
    if (kt + 32 < N_) {
      stage(1 - p, kt + 32);
#if HAVE_TDM
      __builtin_amdgcn_s_wait_tensorcnt(1);   // this iter's tile complete
#endif
    } else {
#if HAVE_TDM
      __builtin_amdgcn_s_wait_tensorcnt(0);
#endif
    }
    __syncthreads();                          // publish buf[p] to all waves

    const __bf16* kb = &KbufS[p][0][0];
    const __bf16* vb = &VbufS[p][0][0];

    // ---- S = Q K^T  (two 16x16 col tiles, K-dim 64) from LDS
    v8f s0 = {}, s1 = {};
    s0 = wmma_bf16(aQ0, *(const v16bf*)(kb + ln * 64 + hf * 16), s0);
    s0 = wmma_bf16(aQ1, *(const v16bf*)(kb + ln * 64 + 32 + hf * 16), s0);
    s1 = wmma_bf16(aQ0, *(const v16bf*)(kb + (16 + ln) * 64 + hf * 16), s1);
    s1 = wmma_bf16(aQ1, *(const v16bf*)(kb + (16 + ln) * 64 + 32 + hf * 16), s1);

    // ---- online softmax in C-frag layout
    float e0[8], e1[8];
#pragma unroll
    for (int r = 0; r < 8; ++r) {
      float mx = fmaxf(s0[r], s1[r]);
      mx = fmaxf(mx, __shfl_xor(mx, 1, 32));
      mx = fmaxf(mx, __shfl_xor(mx, 2, 32));
      mx = fmaxf(mx, __shfl_xor(mx, 4, 32));
      mx = fmaxf(mx, __shfl_xor(mx, 8, 32));
      float mn = fmaxf(mrow[r], mx);
      float al = __expf(mrow[r] - mn);
      float a0 = __expf(s0[r] - mn);
      float a1 = __expf(s1[r] - mn);
      float rs = a0 + a1;
      rs += __shfl_xor(rs, 1, 32);
      rs += __shfl_xor(rs, 2, 32);
      rs += __shfl_xor(rs, 4, 32);
      rs += __shfl_xor(rs, 8, 32);
      lrow[r] = lrow[r] * al + rs;
      mrow[r] = mn;
      o0[r] *= al; o1[r] *= al; o2[r] *= al; o3[r] *= al;
      e0[r] = a0; e1[r] = a1;
    }

    // ---- P: C-frag -> A-frag via per-wave LDS tile
#pragma unroll
    for (int r = 0; r < 8; ++r) {
      int row = r + hf * 8;
      pp[row * 32 + ln]      = (__bf16)e0[r];
      pp[row * 32 + 16 + ln] = (__bf16)e1[r];
    }
    asm volatile("s_wait_dscnt 0" ::: "memory");
    v16bf aP = make_a(*(const v8bf*)(pp + ln * 32 + hf * 16),
                      *(const v8bf*)(pp + ln * 32 + 8 + hf * 16));

    // ---- O += P * V from LDS (V^T tile: contiguous B-fragments)
    o0 = wmma_bf16(aP, *(const v16bf*)(vb + ln * 32 + hf * 16), o0);
    o1 = wmma_bf16(aP, *(const v16bf*)(vb + (16 + ln) * 32 + hf * 16), o1);
    o2 = wmma_bf16(aP, *(const v16bf*)(vb + (32 + ln) * 32 + hf * 16), o2);
    o3 = wmma_bf16(aP, *(const v16bf*)(vb + (48 + ln) * 32 + hf * 16), o3);

    __syncthreads();                          // done reading buf[p]
    p ^= 1;
  }

  // ---- normalize + write [B,N,H*D] bf16
#pragma unroll
  for (int r = 0; r < 8; ++r) {
    float inv = 1.0f / lrow[r];
    int nq = nq0 + r + hf * 8;
    size_t base = ((size_t)(b * N_ + nq)) * C_ + h * D_;
    attn_out[base + ln]      = (__bf16)(o0[r] * inv);
    attn_out[base + 16 + ln] = (__bf16)(o1[r] * inv);
    attn_out[base + 32 + ln] = (__bf16)(o2[r] * inv);
    attn_out[base + 48 + ln] = (__bf16)(o3[r] * inv);
  }
}

// ---------------------------------------------------------------- proj GEMM
__global__ void __launch_bounds__(128)
k_proj_gemm(const __bf16* __restrict__ ab, const __bf16* __restrict__ wT,
            const float* __restrict__ bias, float* __restrict__ out) {
  const int lane = threadIdx.x & 31;
  const int wave = threadIdx.x >> 5;
  const int ln = lane & 15, hf = lane >> 4;
  const int NTB = C_ / 64;                    // 16 column blocks
  int wtile = blockIdx.x * 4 + wave;          // 0 .. 64*16-1
  int ntb = wtile % NTB, mtb = wtile / NTB;

  const __bf16* ap0 = ab + (size_t)(mtb * 64 + ln) * C_ + hf * 16;
  const __bf16* bp0 = wT + (size_t)(ntb * 64 + ln) * C_ + hf * 16;

  v8f acc[4][4];
#pragma unroll
  for (int i = 0; i < 4; ++i)
#pragma unroll
    for (int j = 0; j < 4; ++j) acc[i][j] = v8f{};

  for (int k0 = 0; k0 < C_; k0 += 32) {
    v16bf a[4], b[4];
#pragma unroll
    for (int i = 0; i < 4; ++i)
      a[i] = make_a(*(const v8bf*)(ap0 + (size_t)i * 16 * C_ + k0),
                    *(const v8bf*)(ap0 + (size_t)i * 16 * C_ + k0 + 8));
#pragma unroll
    for (int j = 0; j < 4; ++j)
      b[j] = *(const v16bf*)(bp0 + (size_t)j * 16 * C_ + k0);
#pragma unroll
    for (int i = 0; i < 4; ++i)
#pragma unroll
      for (int j = 0; j < 4; ++j) acc[i][j] = wmma_bf16(a[i], b[j], acc[i][j]);
  }

#pragma unroll
  for (int j = 0; j < 4; ++j) {
    int col = ntb * 64 + j * 16 + ln;
    float bj = bias[col];
#pragma unroll
    for (int i = 0; i < 4; ++i)
#pragma unroll
      for (int r = 0; r < 8; ++r) {
        int t = mtb * 64 + i * 16 + r + hf * 8;
        out[(size_t)t * C_ + col] = acc[i][j][r] + bj;
      }
  }
}

// ---------------------------------------------------------------- launcher
extern "C" void kernel_launch(void* const* d_in, const int* in_sizes, int n_in,
                              void* d_out, int out_size, void* d_ws,
                              size_t ws_size, hipStream_t stream) {
  const float* x      = (const float*)d_in[0];  // [B,N,C]
  const float* w_qkv  = (const float*)d_in[1];  // [C,3C]
  const float* w_proj = (const float*)d_in[2];  // [C,C]
  const float* b_proj = (const float*)d_in[3];  // [C]
  float* out = (float*)d_out;

  __bf16* ws = (__bf16*)d_ws;
  size_t off = 0;
  __bf16* xb     = ws + off; off += (size_t)B_ * N_ * C_;
  __bf16* wqkvT  = ws + off; off += (size_t)3 * C_ * C_;
  __bf16* wprojT = ws + off; off += (size_t)C_ * C_;
  __bf16* Qb     = ws + off; off += (size_t)B_ * H_ * N_ * D_;
  __bf16* Kb     = ws + off; off += (size_t)B_ * H_ * N_ * D_;
  __bf16* Vt     = ws + off; off += (size_t)B_ * H_ * N_ * D_;
  __bf16* ao     = ws + off; off += (size_t)B_ * N_ * C_;      // ~48 MB total

  k_convert<<<(B_ * N_ * C_) / 256, 256, 0, stream>>>(x, xb, B_ * N_ * C_);
  k_transpose_cvt<<<(C_ * 3 * C_) / 256, 256, 0, stream>>>(w_qkv, wqkvT, C_,
                                                           3 * C_);
  k_transpose_cvt<<<(C_ * C_) / 256, 256, 0, stream>>>(w_proj, wprojT, C_, C_);

  // QKV: 64x48 64-blocks, 4 waves/block
  k_qkv_gemm<<<(64 * 48) / 4, 128, 0, stream>>>(xb, wqkvT, Qb, Kb, Vt);
  // attention: B*H * 32 query-blocks of 64
  k_attention<<<B_ * H_ * 32, 128, 0, stream>>>(Qb, Kb, Vt, ao);
  // proj: 64x16 64-blocks, 4 waves/block
  k_proj_gemm<<<(64 * 16) / 4, 128, 0, stream>>>(ao, wprojT, b_proj, out);
}
